// TokenMergeLayer_40913858461804
// MI455X (gfx1250) — matrix operand order repaired and checked by reading, not compiled, not run on hardware
//
#include <hip/hip_runtime.h>
#include <hip/hip_bf16.h>
#include <math.h>

// ---------------------------------------------------------------------------
// TokenMergeLayer for MI455X (gfx1250, wave32, WMMA).
// Dominant cost: per-(b,i) GEMM (384x128)@(128x256) of the antisymmetric
// Hadamard direction term -> v_wmma_f32_16x16x32_bf16 (f32 accumulate).
// Sigmoids use v_rcp_f32 (1 ulp) instead of the IEEE divide expansion to keep
// the VALU stream short so it co-executes under the WMMA (XDL) issue slots.
// ---------------------------------------------------------------------------

typedef __bf16 bf16_t;
typedef bf16_t v16bf __attribute__((ext_vector_type(16)));
typedef bf16_t v8bf  __attribute__((ext_vector_type(8)));
typedef float  v8f   __attribute__((ext_vector_type(8)));

constexpr int B_    = 2;
constexpr int N_    = 384;
constexpr int D_    = 256;
constexpr int DDIR_ = 128;
constexpr int DSIM_ = 64;
constexpr int ROWS_ = B_ * N_;           // 768
constexpr float EPS_   = 1e-6f;
constexpr float DELTA_ = 1e-6f;

// workspace layout (in floats)
constexpr int OFF_Q  = 0;
constexpr int OFF_K  = OFF_Q  + ROWS_ * DSIM_;     // 49152
constexpr int OFF_QD = OFF_K  + ROWS_ * DSIM_;     // 98304
constexpr int OFF_KD = OFF_QD + ROWS_ * DDIR_;     // 196608
constexpr int OFF_S  = OFF_KD + ROWS_ * DDIR_;     // 294912
constexpr int OFF_A  = OFF_S  + B_ * N_ * N_;      // 589824
constexpr int OFF_M  = OFF_A  + B_ * N_ * N_;      // 884736
constexpr int OFF_ST = OFF_M  + ROWS_;             // 885504 (4 floats)
constexpr int OFF_VF = ((OFF_ST + 4 + 7) / 8) * 8; // 885512 (32B aligned, bf16 frag Vd)

// fast sigmoid: v_exp_f32 + v_rcp_f32 (gate tolerance >> 1 ulp)
__device__ __forceinline__ float sigmoidf_(float x) {
  return __builtin_amdgcn_rcpf(1.0f + __expf(-x));
}

// --- k0: Vd (f32 row-major 128x256) -> bf16 in WMMA B-fragment order -------
// frag element (d_tile, kc, lane, h): lane<16 -> col d = 16*dt + lane, K = 32*kc + h
//                                     lane>=16 -> col d = 16*dt + lane-16, K = 32*kc + 16 + h
__global__ void k0_vdfrag(const float* __restrict__ Vd, bf16_t* __restrict__ VdFrag) {
  int e = blockIdx.x * 256 + threadIdx.x;
  if (e >= DDIR_ * D_) return;
  int r = e >> 8;          // K index 0..127  (row-major read: coalesced in d)
  int d = e & 255;         // col 0..255
  int dt = d >> 4, kc = r >> 5;
  int lane = (((r & 31) >= 16) ? 16 : 0) + (d & 15);
  int h = r & 15;
  int dst = (((dt << 2) + kc) * 32 + lane) * 16 + h;
  VdFrag[dst] = (bf16_t)Vd[e];
}

// --- k1: projections q,k (d=64), qd,kd (d=128): one block per token row ----
__global__ void k1_proj(const float* __restrict__ t,
                        const float* __restrict__ Qd, const float* __restrict__ Kd,
                        const float* __restrict__ Qs, const float* __restrict__ Ks,
                        float* __restrict__ q, float* __restrict__ k,
                        float* __restrict__ qd, float* __restrict__ kd) {
  __shared__ float trow[D_];
  int row = blockIdx.x, tid = threadIdx.x;
  trow[tid] = t[row * D_ + tid];
  __syncthreads();
  for (int o = tid; o < 2 * DSIM_ + 2 * DDIR_; o += 256) {
    const float* W; float* dst; int cols, c;
    if (o < DSIM_)                { W = Qs; dst = q;  cols = DSIM_; c = o; }
    else if (o < 2 * DSIM_)       { W = Ks; dst = k;  cols = DSIM_; c = o - DSIM_; }
    else if (o < 2*DSIM_ + DDIR_) { W = Qd; dst = qd; cols = DDIR_; c = o - 2*DSIM_; }
    else                          { W = Kd; dst = kd; cols = DDIR_; c = o - 2*DSIM_ - DDIR_; }
    float acc = 0.f;
    for (int r = 0; r < D_; ++r) acc += trow[r] * W[r * cols + c];
    dst[row * cols + c] = acc;
  }
}

// --- k2: symmetric similarity s = 0.5*(qk^T + kq^T)/sqrt(64) ---------------
__global__ void k2_sim(const float* __restrict__ q, const float* __restrict__ k,
                       float* __restrict__ s) {
  __shared__ float qI[16][DSIM_], kJ[16][DSIM_], qJ[16][DSIM_], kI[16][DSIM_];
  int bb = blockIdx.x;
  int b = bb / (24 * 24), rem = bb % (24 * 24);
  int it = rem / 24, jt = rem % 24;
  int tid = threadIdx.x;
  for (int e = tid; e < 16 * DSIM_; e += 256) {
    int rl = e >> 6, c = e & 63;
    qI[rl][c] = q[(b*N_ + it*16 + rl) * DSIM_ + c];
    kI[rl][c] = k[(b*N_ + it*16 + rl) * DSIM_ + c];
    qJ[rl][c] = q[(b*N_ + jt*16 + rl) * DSIM_ + c];
    kJ[rl][c] = k[(b*N_ + jt*16 + rl) * DSIM_ + c];
  }
  __syncthreads();
  int il = tid >> 4, jl = tid & 15;
  float acc = 0.f;
  for (int c = 0; c < DSIM_; ++c)
    acc += qI[il][c] * kJ[jl][c] + qJ[jl][c] * kI[il][c];
  s[(b*N_ + it*16 + il) * N_ + jt*16 + jl] = acc * 0.0625f; // 0.5 / sqrt(64)
}

// --- k2b: per-batch off-diagonal mean + rstd -------------------------------
__global__ void k2b_stats(const float* __restrict__ s, float* __restrict__ stats) {
  __shared__ float rs[256], rq[256];
  int b = blockIdx.x, tid = threadIdx.x;
  float sum = 0.f, sq = 0.f;
  for (int e = tid; e < N_ * N_; e += 256) {
    int i = e / N_, j = e - i * N_;
    if (i != j) { float v = s[b*N_*N_ + e]; sum += v; sq += v * v; }
  }
  rs[tid] = sum; rq[tid] = sq; __syncthreads();
  for (int st = 128; st > 0; st >>= 1) {
    if (tid < st) { rs[tid] += rs[tid + st]; rq[tid] += rq[tid + st]; }
    __syncthreads();
  }
  if (tid == 0) {
    float cnt = (float)(N_ * (N_ - 1));
    float mu = rs[0] / cnt;
    float var = rq[0] / cnt - mu * mu;
    stats[b * 2]     = mu;
    stats[b * 2 + 1] = rsqrtf(var + EPS_);
  }
}

// --- k3a: masses m_i = 1/(sum_j exp(tilde_s[i,j]) + eps) -------------------
__global__ void k3a_m(const float* __restrict__ s, const float* __restrict__ stats,
                      const float* __restrict__ a_l, const float* __restrict__ b_l,
                      float* __restrict__ m) {
  __shared__ float red[256];
  int row = blockIdx.x, tid = threadIdx.x;
  int b = row / N_;
  float a = a_l[0], bb = b_l[0], mu = stats[b*2], rstd = stats[b*2+1];
  float e = 0.f;
  for (int j = tid; j < N_; j += 256) {
    float tl = a * (s[row*N_ + j] - mu) * rstd + bb;
    e += __expf(tl);                  // GAMMA = 1
  }
  red[tid] = e; __syncthreads();
  for (int st = 128; st > 0; st >>= 1) { if (tid < st) red[tid] += red[tid + st]; __syncthreads(); }
  if (tid == 0) m[row] = 1.0f / (red[0] + EPS_);   // cold path: keep exact divide
}

// --- k3b: alpha_ij = A_ij^2 m_j / (sum_j A_ij m_j + delta), diag -> 0 ------
__global__ void k3b_alpha(const float* __restrict__ s, const float* __restrict__ stats,
                          const float* __restrict__ a_l, const float* __restrict__ b_l,
                          const float* __restrict__ m, float* __restrict__ alpha) {
  __shared__ float red[256];
  __shared__ float s2s;
  int row = blockIdx.x, tid = threadIdx.x;
  int b = row / N_, i = row % N_;
  float a = a_l[0], bb = b_l[0], mu = stats[b*2], rstd = stats[b*2+1];
  float s2 = 0.f;
  for (int j = tid; j < N_; j += 256) {
    float tl = a * (s[row*N_ + j] - mu) * rstd + bb;
    float A = sigmoidf_(tl);
    s2 += A * m[b*N_ + j];
  }
  red[tid] = s2; __syncthreads();
  for (int st = 128; st > 0; st >>= 1) { if (tid < st) red[tid] += red[tid + st]; __syncthreads(); }
  if (tid == 0) s2s = 1.0f / (red[0] + DELTA_);    // one exact divide per row
  __syncthreads();
  float inv = s2s;
  for (int j = tid; j < N_; j += 256) {
    float tl = a * (s[row*N_ + j] - mu) * rstd + bb;
    float A = sigmoidf_(tl);
    alpha[row*N_ + j] = (j == i) ? 0.f : A * A * m[b*N_ + j] * inv;
  }
}

// --- k4: the big one. One block per (b,i): for all j,
//   delta[j][d] = sum_r (qd_i[r]*kd_j[r] - qd_j[r]*kd_i[r]) * Vd[r][d]   (WMMA bf16)
//   t_next_i[d] = t_i[d] - sum_j alpha_ij * sigmoid(delta) * (t_i[d]-t_j[d])
__global__ void __launch_bounds__(256) k4_update(
    const float* __restrict__ t,
    const float* __restrict__ qd, const float* __restrict__ kd,
    const bf16_t* __restrict__ VdFrag,
    const float* __restrict__ alpha,
    float* __restrict__ out) {
  __shared__ bf16_t AL[16][136];         // 16 j-rows x 128 K, +8 pad halves -> bank rotate
  __shared__ float qdI[DDIR_], kdI[DDIR_], tI[D_], alphaR[N_];
  __shared__ float tJ[16 * 260];         // 16 j-rows x 256 d, stride 260 avoids conflicts

  int row = blockIdx.x, tid = threadIdx.x;
  int b = row / N_;
  int wave = tid >> 5, lane = tid & 31, half = lane >> 4, ln = lane & 15;

  if (tid < DDIR_) { qdI[tid] = qd[row*DDIR_ + tid]; kdI[tid] = kd[row*DDIR_ + tid]; }
  tI[tid] = t[row*D_ + tid];
  for (int e = tid; e < N_; e += 256) alphaR[e] = alpha[row*N_ + e];

  int dt0 = wave * 2;                     // each wave owns 2 of 16 d-tiles
  float acc0 = 0.f, acc1 = 0.f;
  int aj = tid >> 4;                      // j for A staging
  int rb = (tid & 15) * 8;                // K base for A staging

  for (int jt = 0; jt < N_ / 16; ++jt) {
    __syncthreads();                      // covers init staging + prev tile consumers
    { // stage A = antisymmetric Hadamard (bf16)
      int jg = jt * 16 + aj;
      const float* qdJ = qd + (b*N_ + jg) * DDIR_;
      const float* kdJ = kd + (b*N_ + jg) * DDIR_;
      #pragma unroll
      for (int u = 0; u < 8; ++u) {
        int r = rb + u;
        AL[aj][r] = (bf16_t)(qdI[r] * kdJ[r] - qdJ[r] * kdI[r]);
      }
    }
    // stage t_j tile
    for (int e = tid; e < 16 * D_; e += 256) {
      int j = e >> 8, d = e & 255;
      tJ[j * 260 + d] = t[(b*N_ + jt*16 + j) * D_ + d];
    }
    __syncthreads();

    #pragma unroll
    for (int h2 = 0; h2 < 2; ++h2) {
      int dt = dt0 + h2;
      v8f c = {0.f, 0.f, 0.f, 0.f, 0.f, 0.f, 0.f, 0.f};
      #pragma unroll
      for (int kc = 0; kc < 4; ++kc) {    // K = 128 in 4 chunks of 32
        // A fragment (contiguous K order, same as VdFrag B): row = ln,
        //   halves 0..7  = K kc*32 + half*16 .. +7
        //   halves 8..15 = K kc*32 + half*16 + 8 .. +7
        int c0 = kc * 32 + half * 16;
        union { v16bf v; v8bf h[2]; } au;
        au.h[0] = *(const v8bf*)&AL[ln][c0];
        au.h[1] = *(const v8bf*)&AL[ln][c0 + 8];
        // B fragment: pre-swizzled, loop-invariant -> compiler keeps in VGPRs
        v16bf bfr = *(const v16bf*)(VdFrag + (((dt * 4 + kc) * 32 + lane) << 4));
        c = __builtin_amdgcn_wmma_f32_16x16x32_bf16(
                false, au.v, false, bfr, (short)0, c, false, false);
      }
      // C/D layout: lane -> col n = ln, VGPR v -> row m = v + 8*half
      int d = dt * 16 + ln;
      float td = tI[d];
      float accl = 0.f;
      #pragma unroll
      for (int v = 0; v < 8; ++v) {
        int jl = v + half * 8;
        float aij = alphaR[jt * 16 + jl];           // zero on diagonal
        float g = sigmoidf_(c[v]);                  // v_exp_f32 + v_rcp_f32
        accl += aij * g * (td - tJ[jl * 260 + d]);
      }
      if (h2 == 0) acc0 += accl; else acc1 += accl;
    }
  }
  // combine the two half-wave partial sums (rows 0-7 vs 8-15 of each j-tile)
  float o0 = acc0 + __shfl_down(acc0, 16, 32);
  float o1 = acc1 + __shfl_down(acc1, 16, 32);
  if (half == 0) {
    out[row*D_ + dt0*16 + ln]       = tI[dt0*16 + ln]       - o0;
    out[row*D_ + (dt0+1)*16 + ln]   = tI[(dt0+1)*16 + ln]   - o1;
  }
}

// ---------------------------------------------------------------------------
extern "C" void kernel_launch(void* const* d_in, const int* in_sizes, int n_in,
                              void* d_out, int out_size, void* d_ws, size_t ws_size,
                              hipStream_t stream) {
  const float* t   = (const float*)d_in[0];
  const float* Qd  = (const float*)d_in[1];
  const float* Kd  = (const float*)d_in[2];
  const float* Vd  = (const float*)d_in[3];
  const float* Qs  = (const float*)d_in[4];
  const float* Ks  = (const float*)d_in[5];
  const float* a_l = (const float*)d_in[6];
  const float* b_l = (const float*)d_in[7];
  float* out = (float*)d_out;
  float* ws  = (float*)d_ws;

  float*  q      = ws + OFF_Q;
  float*  k      = ws + OFF_K;
  float*  qd     = ws + OFF_QD;
  float*  kd     = ws + OFF_KD;
  float*  s      = ws + OFF_S;
  float*  alpha  = ws + OFF_A;
  float*  m      = ws + OFF_M;
  float*  stats  = ws + OFF_ST;
  bf16_t* VdFrag = (bf16_t*)(ws + OFF_VF);

  k0_vdfrag<<<(DDIR_ * D_ + 255) / 256, 256, 0, stream>>>(Vd, VdFrag);
  k1_proj  <<<ROWS_, 256, 0, stream>>>(t, Qd, Kd, Qs, Ks, q, k, qd, kd);
  k2_sim   <<<B_ * (N_/16) * (N_/16), 256, 0, stream>>>(q, k, s);
  k2b_stats<<<B_, 256, 0, stream>>>(s, stats);
  k3a_m    <<<ROWS_, 256, 0, stream>>>(s, stats, a_l, b_l, m);
  k3b_alpha<<<ROWS_, 256, 0, stream>>>(s, stats, a_l, b_l, m, alpha);
  k4_update<<<ROWS_, 256, 0, stream>>>(t, qd, kd, VdFrag, alpha, out);
}
